// FlashInferSparseMoeBlock_89446988906794
// MI455X (gfx1250) — matrix operand order repaired and checked by reading, not compiled, not run on hardware
//
#include <hip/hip_runtime.h>
#include <cstdint>
#include <cstddef>

// ---------------------------------------------------------------------------
// FlashInfer-style sparse MoE block for MI455X (gfx1250, wave32).
// fp32 storage, bf16 WMMA compute (v_wmma_f32_16x16x32_bf16).
// Pipeline per K-chunk: async global->LDS DMA (ASYNCcnt) into thread-private
// f32 landing zones -> v_cvt_pk_bf16_f32 into double-buffered shared bf16
// tiles -> ds_load_b128 fragments -> WMMA. One barrier per chunk.
// Top-2 tile skip; NT temporal hint on streamed (read-once) weights.
// ---------------------------------------------------------------------------

#define T_TOK 8192
#define HDIM  2048
#define IDIM  1408
#define NEXP  8

typedef __attribute__((ext_vector_type(16))) __bf16 v16bf;
typedef __attribute__((ext_vector_type(4)))  __bf16 v4bf;
typedef __attribute__((ext_vector_type(8)))  float  v8f;
typedef __attribute__((ext_vector_type(4)))  float  v4fv;

struct alignas(16) F4 { float x, y, z, w; };
struct alignas(16) Q4 { unsigned int a, b, c, d; };

union Frag32B {
    Q4    q[2];
    v16bf v;
};

// round-to-nearest-even f32 -> bf16 bits (epilogue scalar use only)
__device__ __forceinline__ unsigned int f2bf1(float f) {
    unsigned int u = __float_as_uint(f);
    return (u + 0x7FFFu + ((u >> 16) & 1u)) >> 16;
}

// 32-bit LDS byte offset of a __shared__ object (generic ptr low bits = LDS addr)
__device__ __forceinline__ uint32_t lds_addr(const void* p) {
    return (uint32_t)(uintptr_t)p;
}

// ---- CDNA5 async global->LDS DMA (ASYNCcnt) -------------------------------
// 64 contiguous bytes per lane. INST_OFFSET applies to both LDS and global
// addresses, so one base pair serves all four b128 ops.
__device__ __forceinline__ void async_copy64_rt(uint32_t lds, const void* g) {
    asm volatile(
        "global_load_async_to_lds_b128 %0, %1, off\n\t"
        "global_load_async_to_lds_b128 %0, %1, off offset:16\n\t"
        "global_load_async_to_lds_b128 %0, %1, off offset:32\n\t"
        "global_load_async_to_lds_b128 %0, %1, off offset:48"
        :: "v"(lds), "v"(g) : "memory");
}
__device__ __forceinline__ void async_copy64_nt(uint32_t lds, const void* g) {
    asm volatile(
        "global_load_async_to_lds_b128 %0, %1, off th:TH_LOAD_NT\n\t"
        "global_load_async_to_lds_b128 %0, %1, off offset:16 th:TH_LOAD_NT\n\t"
        "global_load_async_to_lds_b128 %0, %1, off offset:32 th:TH_LOAD_NT\n\t"
        "global_load_async_to_lds_b128 %0, %1, off offset:48 th:TH_LOAD_NT"
        :: "v"(lds), "v"(g) : "memory");
}
// 32 contiguous bytes per lane (bf16 h tile row-half)
__device__ __forceinline__ void async_copy32_rt(uint32_t lds, const void* g) {
    asm volatile(
        "global_load_async_to_lds_b128 %0, %1, off\n\t"
        "global_load_async_to_lds_b128 %0, %1, off offset:16"
        :: "v"(lds), "v"(g) : "memory");
}
__device__ __forceinline__ void wait_async0() {
    asm volatile("s_wait_asynccnt 0" ::: "memory");
}

// ---- f32 -> bf16 tile conversion (hardware v_cvt_pk_bf16_f32) -------------
// Converts the 16 floats this thread DMA'd (thread-private region).
__device__ __forceinline__ void cvt_row16(const float* __restrict__ srcRow,
                                          unsigned short* __restrict__ dstRow) {
    const v4fv* p = (const v4fv*)srcRow;
    v4fv a0 = p[0], a1 = p[1], a2 = p[2], a3 = p[3];
    v4bf b0 = __builtin_convertvector(a0, v4bf);
    v4bf b1 = __builtin_convertvector(a1, v4bf);
    v4bf b2 = __builtin_convertvector(a2, v4bf);
    v4bf b3 = __builtin_convertvector(a3, v4bf);
    v4bf* d = (v4bf*)dstRow;
    d[0] = b0; d[1] = b1; d[2] = b2; d[3] = b3;
}

// ---- fragment loaders (bf16 LDS tiles, pure b128 reads) -------------------
// A-fragment (16x32, MxK):
// lane<16: row=lane, K={0..7,16..23}; lane>=16: row=lane-16, K={8..15,24..31}
__device__ __forceinline__ v16bf load_a_frag(const unsigned short (*As)[40],
                                             int rowBase, int lane) {
    const int row = rowBase + (lane & 15);
    const int kb  = (lane >> 4) << 3;           // 0 or 8
    Frag32B f;
    f.q[0] = *(const Q4*)&As[row][kb];
    f.q[1] = *(const Q4*)&As[row][kb + 16];
    return f.v;
}
// B-fragment (32x16, KxN), LDS holds W[n][k] (B^T rows):
// lane<16: N=lane, K={0..7,16..23} ; lane>=16: N=lane-16, K={8..15,24..31}
__device__ __forceinline__ v16bf load_b_frag(const unsigned short (*Bs)[40],
                                             int colBase, int lane) {
    const int col = colBase + (lane & 15);
    const int kb  = (lane >> 4) << 3;           // 0 or 8
    Frag32B f;
    f.q[0] = *(const Q4*)&Bs[col][kb];
    f.q[1] = *(const Q4*)&Bs[col][kb + 16];
    return f.v;
}

// ---------------------------------------------------------------------------
// Output zero-init
// ---------------------------------------------------------------------------
__global__ void zero_f4(F4* __restrict__ p, int n) {
    int i = blockIdx.x * blockDim.x + threadIdx.x;
    if (i < n) p[i] = F4{0.0f, 0.0f, 0.0f, 0.0f};
}

// ---------------------------------------------------------------------------
// Router: one wave per token; butterfly shuffle reduce; softmax; top-2;
// renormalize. Writes raw logits (d_out tail) + dense combine[T,8] (ws).
// ---------------------------------------------------------------------------
__global__ __launch_bounds__(256)
void router_kernel(const float* __restrict__ x,
                   const float* __restrict__ rw,
                   float* __restrict__ logits_out,
                   float* __restrict__ combine) {
    const int lane = threadIdx.x & 31;
    const int wv   = threadIdx.x >> 5;
    const int t    = blockIdx.x * 8 + wv;

    float acc[NEXP];
#pragma unroll
    for (int e = 0; e < NEXP; ++e) acc[e] = 0.0f;

    const float* xr = x + (size_t)t * HDIM;
    for (int k = lane; k < HDIM; k += 32) {
        float xv = xr[k];
#pragma unroll
        for (int e = 0; e < NEXP; ++e) acc[e] += xv * rw[e * HDIM + k];
    }
#pragma unroll
    for (int e = 0; e < NEXP; ++e) {
#pragma unroll
        for (int off = 16; off > 0; off >>= 1)
            acc[e] += __shfl_xor(acc[e], off, 32);
    }

    if (lane == 0) {
        float m = acc[0];
#pragma unroll
        for (int e = 1; e < NEXP; ++e) m = fmaxf(m, acc[e]);
        float p[NEXP], s = 0.0f;
#pragma unroll
        for (int e = 0; e < NEXP; ++e) { p[e] = __expf(acc[e] - m); s += p[e]; }
        float inv = 1.0f / s;
#pragma unroll
        for (int e = 0; e < NEXP; ++e) p[e] *= inv;

        int i1 = 0;
#pragma unroll
        for (int e = 1; e < NEXP; ++e) if (p[e] > p[i1]) i1 = e;
        int i2 = (i1 == 0) ? 1 : 0;
#pragma unroll
        for (int e = 0; e < NEXP; ++e) if (e != i1 && p[e] > p[i2]) i2 = e;

        float d  = p[i1] + p[i2];
        float w1 = p[i1] / d, w2 = p[i2] / d;
#pragma unroll
        for (int e = 0; e < NEXP; ++e) {
            logits_out[(size_t)t * NEXP + e] = acc[e];
            combine[(size_t)t * NEXP + e] = (e == i1) ? w1 : ((e == i2) ? w2 : 0.0f);
        }
    }
}

// ---------------------------------------------------------------------------
// Expert stage 1: h = silu(x @ gate^T) * (x @ up^T) -> bf16 workspace.
// ---------------------------------------------------------------------------
__global__ __launch_bounds__(256)
void moe_gateup_kernel(const float* __restrict__ x,
                       const float* __restrict__ gate_w,
                       const float* __restrict__ up_w,
                       const float* __restrict__ combine,
                       unsigned short* __restrict__ hbuf,
                       int expert) {
    // thread-private f32 DMA landing zones (no barriers needed)
    __shared__ alignas(16) float Axf[128][36];
    __shared__ alignas(16) float Bgf[128][36];
    __shared__ alignas(16) float Buf_[128][36];
    // double-buffered shared bf16 tiles (fragment source)
    __shared__ alignas(16) unsigned short Ax16[2][128][40];
    __shared__ alignas(16) unsigned short Bg16[2][128][40];
    __shared__ alignas(16) unsigned short Bu16[2][128][40];
    __shared__ int flag;

    const int tid  = threadIdx.x;
    const int lane = tid & 31;
    const int wv   = tid >> 5;
    const int m0   = blockIdx.y * 128;
    const int n0   = blockIdx.x * 128;

    if (tid == 0) flag = 0;
    __syncthreads();
    if (tid < 128) {
        if (combine[(size_t)(m0 + tid) * NEXP + expert] != 0.0f) flag = 1;
    }
    __syncthreads();
    if (!flag) return;                 // no token in this tile routed to expert

    const float* gw = gate_w + (size_t)expert * IDIM * HDIM;
    const float* uw = up_w   + (size_t)expert * IDIM * HDIM;

    // staging: 2 threads/row, 64B (16 floats) per thread per tile per K-chunk
    const int r  = tid >> 1;
    const int cf = (tid & 1) << 4;
    const float* gxr = x  + (size_t)(m0 + r) * HDIM + cf;
    const float* ggr = gw + (size_t)(n0 + r) * HDIM + cf;
    const float* gur = uw + (size_t)(n0 + r) * HDIM + cf;
    const uint32_t lAx = lds_addr(&Axf[r][cf]);
    const uint32_t lBg = lds_addr(&Bgf[r][cf]);
    const uint32_t lBu = lds_addr(&Buf_[r][cf]);

    // prologue
    async_copy64_rt(lAx, gxr);
    async_copy64_nt(lBg, ggr);
    async_copy64_nt(lBu, gur);

    const v8f zf = {};
    v8f accG[8], accU[8];
#pragma unroll
    for (int j = 0; j < 8; ++j) { accG[j] = zf; accU[j] = zf; }

    const int NK = HDIM / 32;
    for (int kc = 0; kc < NK; ++kc) {
        const int cur = kc & 1;
        wait_async0();                               // own DMA(kc) landed
        cvt_row16(&Axf[r][cf],  &Ax16[cur][r][cf]);  // private bytes -> shared bf16
        cvt_row16(&Bgf[r][cf],  &Bg16[cur][r][cf]);
        cvt_row16(&Buf_[r][cf], &Bu16[cur][r][cf]);
        if (kc + 1 < NK) {
            const int k0n = (kc + 1) * 32;
            async_copy64_rt(lAx, gxr + k0n);         // landing zone already consumed
            async_copy64_nt(lBg, ggr + k0n);
            async_copy64_nt(lBu, gur + k0n);
        }
        __syncthreads();                             // publish bf16[cur]

        v16bf a = load_a_frag(Ax16[cur], wv * 16, lane);
#pragma unroll
        for (int j = 0; j < 8; ++j) {
            v16bf bg = load_b_frag(Bg16[cur], j * 16, lane);
            accG[j] = __builtin_amdgcn_wmma_f32_16x16x32_bf16(
                false, a, false, bg, (short)0, accG[j], false, false);
            v16bf bu = load_b_frag(Bu16[cur], j * 16, lane);
            accU[j] = __builtin_amdgcn_wmma_f32_16x16x32_bf16(
                false, a, false, bu, (short)0, accU[j], false, false);
        }
    }

    // fused SwiGLU epilogue; C/D layout: elem i -> M = i + 8*(lane>>4), N = lane&15
    const int hi2 = lane >> 4;
    const int nn  = lane & 15;
#pragma unroll
    for (int j = 0; j < 8; ++j) {
#pragma unroll
        for (int i = 0; i < 8; ++i) {
            float g = accG[j][i];
            float u = accU[j][i];
            float s = g / (1.0f + __expf(-g));
            int tok = m0 + wv * 16 + i + 8 * hi2;
            int col = n0 + j * 16 + nn;
            hbuf[(size_t)tok * IDIM + col] = (unsigned short)f2bf1(s * u);
        }
    }
}

// ---------------------------------------------------------------------------
// Expert stage 2: out += combine[:,e] * (h @ down^T). h is bf16 already:
// DMA straight into the (wave-local) A tile, no conversion.
// ---------------------------------------------------------------------------
__global__ __launch_bounds__(256)
void moe_down_kernel(const unsigned short* __restrict__ hbuf,
                     const float* __restrict__ down_w,
                     const float* __restrict__ combine,
                     float* __restrict__ out,
                     int expert) {
    __shared__ alignas(16) unsigned short Ah[2][128][40];   // DMA target (bf16)
    __shared__ alignas(16) float Bdf[128][36];              // private f32 landing
    __shared__ alignas(16) unsigned short Bd16[2][128][40];
    __shared__ float cs[128];
    __shared__ int flag;

    const int tid  = threadIdx.x;
    const int lane = tid & 31;
    const int wv   = tid >> 5;
    const int m0   = blockIdx.y * 128;
    const int n0   = blockIdx.x * 128;

    if (tid == 0) flag = 0;
    __syncthreads();
    if (tid < 128) {
        float c = combine[(size_t)(m0 + tid) * NEXP + expert];
        cs[tid] = c;
        if (c != 0.0f) flag = 1;
    }
    __syncthreads();
    if (!flag) return;

    const float* dw = down_w + (size_t)expert * HDIM * IDIM;

    const int r  = tid >> 1;
    const int ch = (tid & 1) << 4;     // 16-element offset (shorts or floats)
    const unsigned short* ghr = hbuf + (size_t)(m0 + r) * IDIM + ch;
    const float*          gdr = dw   + (size_t)(n0 + r) * IDIM + ch;
    const uint32_t lH[2] = { lds_addr(&Ah[0][r][ch]), lds_addr(&Ah[1][r][ch]) };
    const uint32_t lBd   = lds_addr(&Bdf[r][ch]);

    async_copy32_rt(lH[0], ghr);
    async_copy64_nt(lBd, gdr);

    const v8f zf = {};
    v8f acc[8];
#pragma unroll
    for (int j = 0; j < 8; ++j) acc[j] = zf;

    const int NK = IDIM / 32;
    for (int kc = 0; kc < NK; ++kc) {
        const int cur = kc & 1;
        wait_async0();
        cvt_row16(&Bdf[r][ch], &Bd16[cur][r][ch]);
        if (kc + 1 < NK) {
            const int nxt = cur ^ 1;
            const int k0n = (kc + 1) * 32;
            async_copy32_rt(lH[nxt], ghr + k0n);
            async_copy64_nt(lBd, gdr + k0n);
        }
        __syncthreads();

        v16bf a = load_a_frag(Ah[cur], wv * 16, lane);
#pragma unroll
        for (int j = 0; j < 8; ++j) {
            v16bf b = load_b_frag(Bd16[cur], j * 16, lane);
            acc[j] = __builtin_amdgcn_wmma_f32_16x16x32_bf16(
                false, a, false, b, (short)0, acc[j], false, false);
        }
    }

    const int hi2 = lane >> 4;
    const int nn  = lane & 15;
#pragma unroll
    for (int j = 0; j < 8; ++j) {
#pragma unroll
        for (int i = 0; i < 8; ++i) {
            int tl  = wv * 16 + i + 8 * hi2;         // local token row
            int tok = m0 + tl;
            int col = n0 + j * 16 + nn;
            float* o = &out[(size_t)tok * HDIM + col];
            *o += cs[tl] * acc[j][i];
        }
    }
}

// ---------------------------------------------------------------------------
// Host launcher
// ---------------------------------------------------------------------------
extern "C" void kernel_launch(void* const* d_in, const int* in_sizes, int n_in,
                              void* d_out, int out_size, void* d_ws, size_t ws_size,
                              hipStream_t stream) {
    (void)in_sizes; (void)n_in; (void)out_size; (void)ws_size;

    const float* x        = (const float*)d_in[0];   // [T, H]
    const float* router_w = (const float*)d_in[1];   // [E, H]
    const float* gate_w   = (const float*)d_in[2];   // [E, I, H]
    const float* up_w     = (const float*)d_in[3];   // [E, I, H]
    const float* down_w   = (const float*)d_in[4];   // [E, H, I]

    float* out    = (float*)d_out;                       // [T, H]
    float* logits = out + (size_t)T_TOK * HDIM;          // [T, E]

    // workspace: combine[T,E] f32, then h[T,I] bf16 (~24 MB total)
    float*          combine = (float*)d_ws;
    unsigned short* hbuf    =
        (unsigned short*)((char*)d_ws + (size_t)T_TOK * NEXP * sizeof(float));

    const int n4 = (T_TOK * HDIM) / 4;
    zero_f4<<<dim3((n4 + 255) / 256), dim3(256), 0, stream>>>((F4*)out, n4);

    router_kernel<<<dim3(T_TOK / 8), dim3(256), 0, stream>>>(
        x, router_w, logits, combine);

    for (int e = 0; e < NEXP; ++e) {
        moe_gateup_kernel<<<dim3(IDIM / 128, T_TOK / 128), dim3(256), 0, stream>>>(
            x, gate_w, up_w, combine, hbuf, e);
        moe_down_kernel<<<dim3(HDIM / 128, T_TOK / 128), dim3(256), 0, stream>>>(
            hbuf, down_w, combine, out, e);
    }
}
